// AttentionBlock_15470472200943
// MI455X (gfx1250) — matrix operand order repaired and checked by reading, not compiled, not run on hardware
//
#include <hip/hip_runtime.h>
#include <stdint.h>

// ---------------------------------------------------------------------------
// Types
// ---------------------------------------------------------------------------
typedef unsigned short u16;
typedef __attribute__((ext_vector_type(16))) __bf16 v16bf;
typedef __attribute__((ext_vector_type(8)))  __bf16 v8bf;
typedef __attribute__((ext_vector_type(8)))  float  v8f;

union Frag { v16bf v; v8bf h[2]; };

#define NB  8      // batch
#define NT  1024   // seq len
#define NE  1024   // embed
#define NH  16     // heads
#define ND  64     // head dim

static __device__ __forceinline__ u16 f2bf(float f) {
  union { float f; unsigned u; } cv; cv.f = f;
  unsigned u = cv.u;
  unsigned r = (u + 0x7FFFu + ((u >> 16) & 1u)) >> 16; // RNE
  return (u16)r;
}

static __device__ __forceinline__ v8f vzero8() {
  v8f z = {0.f, 0.f, 0.f, 0.f, 0.f, 0.f, 0.f, 0.f};
  return z;
}

static __device__ __forceinline__ v8f wmma_bf16(const Frag& a, const Frag& b, v8f c) {
  // (neg_a, A, neg_b, B, c_mod, C, reuse_a, reuse_b)
  return __builtin_amdgcn_wmma_f32_16x16x32_bf16(false, a.v, false, b.v,
                                                 (short)0, c, false, false);
}

// --- CDNA5 async global->LDS DMA (ASYNCcnt path), 16B per lane ---------------
static __device__ __forceinline__ void async_ld16(const u16* gptr, u16* lptr) {
  unsigned l = (unsigned)(size_t)lptr;               // LDS byte offset
  unsigned long long g = (unsigned long long)(size_t)gptr;
  asm volatile("global_load_async_to_lds_b128 %0, %1, off"
               :: "v"(l), "v"(g) : "memory");
}
static __device__ __forceinline__ void wait_async0() {
  asm volatile("s_wait_asynccnt 0x0" ::: "memory");
}

// ---------------------------------------------------------------------------
// Kernel 0: fp32 -> bf16 bulk convert (n multiple of 4)
// ---------------------------------------------------------------------------
__global__ __launch_bounds__(256) void k_cvt_bf16(const float* __restrict__ src,
                                                  u16* __restrict__ dst, int n) {
  int i = (blockIdx.x * 256 + threadIdx.x) * 4;
  if (i < n) {
    float4 f = *(const float4*)(src + i);
    union { u16 b[4]; uint2 u; } pk;
    pk.b[0] = f2bf(f.x); pk.b[1] = f2bf(f.y);
    pk.b[2] = f2bf(f.z); pk.b[3] = f2bf(f.w);
    *(uint2*)(dst + i) = pk.u;
  }
}

// ---------------------------------------------------------------------------
// Kernel 1: QKV GEMM (8192x3072x1024, bf16 WMMA) + bias + RoPE + scatter
//   q,k -> [B,H,T,D] bf16 ; v -> [B,H,D,T] bf16 (transposed for P@V frags)
//   Tiles staged with async global->LDS DMA, double buffered.
// ---------------------------------------------------------------------------
#define LDT 40   // padded LDS stride (elems) for 32-elem K rows (80B, 16B aligned)

__global__ __launch_bounds__(256) void k_qkv_rope(
    const u16* __restrict__ xb,       // [8192,1024] bf16
    const u16* __restrict__ wb,       // [3072,1024] bf16
    const float* __restrict__ bias,   // [3072]
    const float* __restrict__ cost,   // [4095,32]
    const float* __restrict__ sint,   // [4095,32]
    u16* __restrict__ qo, u16* __restrict__ ko, u16* __restrict__ vto) {
  __shared__ __align__(16) u16 At[2][128 * LDT];
  __shared__ __align__(16) u16 Bt[2][128 * LDT];
  const int tid  = threadIdx.x;
  const int lane = tid & 31, wid = tid >> 5;
  const int half = lane >> 4, lm = lane & 15;
  const int wm = wid >> 2, wn = wid & 3;     // 2 x 4 wave grid
  const int m0 = blockIdx.y * 128, n0 = blockIdx.x * 128;

  v8f acc[4][2];
#pragma unroll
  for (int i = 0; i < 4; ++i)
#pragma unroll
    for (int j = 0; j < 2; ++j) acc[i][j] = vzero8();

  // prologue: async prefetch tile k0=0 into buffer 0
#pragma unroll
  for (int p = 0; p < 2; ++p) {
    int id = tid + p * 256;
    int rr = id >> 2, cc = (id & 3) * 8;
    async_ld16(xb + (size_t)(m0 + rr) * 1024 + cc, &At[0][rr * LDT + cc]);
    async_ld16(wb + (size_t)(n0 + rr) * 1024 + cc, &Bt[0][rr * LDT + cc]);
  }
  wait_async0();
  __syncthreads();

  int cur = 0;
  for (int k0 = 0; k0 < 1024; k0 += 32, cur ^= 1) {
    if (k0 + 32 < 1024) {                    // async prefetch next tile
#pragma unroll
      for (int p = 0; p < 2; ++p) {
        int id = tid + p * 256;
        int rr = id >> 2, cc = (id & 3) * 8;
        async_ld16(xb + (size_t)(m0 + rr) * 1024 + k0 + 32 + cc,
                   &At[cur ^ 1][rr * LDT + cc]);
        async_ld16(wb + (size_t)(n0 + rr) * 1024 + k0 + 32 + cc,
                   &Bt[cur ^ 1][rr * LDT + cc]);
      }
    }
    Frag a[4], b[2];
#pragma unroll
    for (int i = 0; i < 4; ++i) {            // A frag: k-halves at +0 / +16
      const u16* base = &At[cur][(wm * 64 + i * 16 + lm) * LDT + half * 8];
      a[i].h[0] = *(const v8bf*)base;
      a[i].h[1] = *(const v8bf*)(base + 16);
    }
#pragma unroll
    for (int j = 0; j < 2; ++j) {            // B frag: k-halves at +0 / +16
      const u16* base = &Bt[cur][(wn * 32 + j * 16 + lm) * LDT + half * 8];
      b[j].h[0] = *(const v8bf*)base;
      b[j].h[1] = *(const v8bf*)(base + 16);
    }
#pragma unroll
    for (int i = 0; i < 4; ++i)
#pragma unroll
      for (int j = 0; j < 2; ++j) acc[i][j] = wmma_bf16(a[i], b[j], acc[i][j]);
    wait_async0();
    __syncthreads();
  }

  // epilogue: bias + RoPE (q,k) + scatter to head layouts
#pragma unroll
  for (int j = 0; j < 2; ++j) {
    const int n = n0 + wn * 32 + j * 16 + lm;  // output feature (0..3071)
    const float bn = bias[n];
    const int sec = n >> 10;                   // 0=q 1=k 2=v (uniform per tile)
    const int idx = n & 1023;
    const int hh = idx >> 6, d = idx & 63;
    const int fr = d >> 1;
#pragma unroll
    for (int i = 0; i < 4; ++i) {
#pragma unroll
      for (int e = 0; e < 8; ++e) {
        const int m = m0 + wm * 64 + i * 16 + e + half * 8;   // token row
        const int bb = m >> 10, t = m & 1023;
        float val = acc[i][j][e] + bn;
        float prt = __shfl_xor(val, 1, 32);    // RoPE partner feature (d^1)
        if (sec < 2) {
          const float cs = cost[t * 32 + fr], sn = sint[t * 32 + fr];
          val = ((d & 1) == 0) ? (val * cs - prt * sn) : (prt * sn + val * cs);
        }
        const size_t bh = (size_t)(bb * NH + hh);
        const u16 ov = f2bf(val);
        if (sec == 0)      qo[(bh * NT + t) * ND + d] = ov;
        else if (sec == 1) ko[(bh * NT + t) * ND + d] = ov;
        else               vto[(bh * ND + d) * NT + t] = ov;
      }
    }
  }
}

// ---------------------------------------------------------------------------
// Kernel 2: causal flash attention, bf16 WMMA, fragment-wise online softmax.
//   grid (B*H, T/128); 8 waves/block, 16 q-rows per wave.
//   K / V^T tiles staged with async global->LDS DMA, double buffered.
// ---------------------------------------------------------------------------
#define KLD 72   // K tile row stride (64 d + pad), 144B, 16B aligned
#define VLD 40   // Vt tile row stride (32 kv + pad)
#define PLD 40   // P tile row stride  (32 kv + pad)

__global__ __launch_bounds__(256) void k_attn(
    const u16* __restrict__ qb, const u16* __restrict__ kb,
    const u16* __restrict__ vtb, u16* __restrict__ yb) {
  __shared__ __align__(16) u16 Kt[2][32 * KLD];
  __shared__ __align__(16) u16 Vt[2][64 * VLD];
  __shared__ __align__(16) u16 Pt[8 * 16 * PLD];
  const int tid  = threadIdx.x;
  const int lane = tid & 31, wid = tid >> 5;
  const int half = lane >> 4, lm = lane & 15;
  const int bh   = blockIdx.x;               // 0..127  (b*NH + h)
  const int qblk = blockIdx.y;               // 0..7
  const int qbase = qblk * 128 + wid * 16;   // wave's first q row
  const size_t base = (size_t)bh * NT * ND;

  // Q fragments held in registers (16 rows x 64 d = 2 k-chunks)
  Frag aq[2];
  {
    const u16* qrow = qb + base + (size_t)(qbase + lm) * ND;
#pragma unroll
    for (int c = 0; c < 2; ++c) {
      aq[c].h[0] = *(const v8bf*)(qrow + c * 32 + half * 8);
      aq[c].h[1] = *(const v8bf*)(qrow + c * 32 + half * 8 + 16);
    }
  }

  v8f o[4];
#pragma unroll
  for (int jd = 0; jd < 4; ++jd) o[jd] = vzero8();
  float mrow[8], lrow[8];
#pragma unroll
  for (int e = 0; e < 8; ++e) { mrow[e] = -1e30f; lrow[e] = 0.f; }

  u16* pw = &Pt[wid * 16 * PLD];             // wave-private P staging
  const int kvend = qblk * 128 + 128;

  // async stage of one 32-kv block into buffer bi
  auto stage = [&](int kv0, int bi) {
    int rr = tid >> 3, cc = (tid & 7) * 8;
    async_ld16(kb + base + (size_t)(kv0 + rr) * ND + cc, &Kt[bi][rr * KLD + cc]);
    int rv = tid >> 2, cv = (tid & 3) * 8;
    async_ld16(vtb + base + (size_t)rv * NT + kv0 + cv, &Vt[bi][rv * VLD + cv]);
  };

  stage(0, 0);
  wait_async0();
  __syncthreads();

  int cur = 0;
  for (int kv0 = 0; kv0 < kvend; kv0 += 32, cur ^= 1) {
    if (kv0 + 32 < kvend) stage(kv0 + 32, cur ^ 1);   // async prefetch next

    if (kv0 <= qbase + 15) {                 // wave-uniform causal skip
      v8f s[2]; s[0] = vzero8(); s[1] = vzero8();
#pragma unroll
      for (int j = 0; j < 2; ++j)
#pragma unroll
        for (int c = 0; c < 2; ++c) {        // B = K^T chunk: lane = kv col
          Frag bk;
          const u16* kp = &Kt[cur][(j * 16 + lm) * KLD + c * 32 + half * 8];
          bk.h[0] = *(const v8bf*)kp;
          bk.h[1] = *(const v8bf*)(kp + 16);
          s[j] = wmma_bf16(aq[c], bk, s[j]);
        }
      // scale + causal mask + online softmax on C-fragment layout
#pragma unroll
      for (int e = 0; e < 8; ++e) {
        const int qr = qbase + e + half * 8;
        float s0 = s[0][e] * 0.125f;
        float s1 = s[1][e] * 0.125f;
        if (kv0 + lm      > qr) s0 = -1e30f;
        if (kv0 + 16 + lm > qr) s1 = -1e30f;
        float rm = fmaxf(s0, s1);
        rm = fmaxf(rm, __shfl_xor(rm, 1, 32));
        rm = fmaxf(rm, __shfl_xor(rm, 2, 32));
        rm = fmaxf(rm, __shfl_xor(rm, 4, 32));
        rm = fmaxf(rm, __shfl_xor(rm, 8, 32));
        const float mn = fmaxf(mrow[e], rm);
        const float al = __expf(mrow[e] - mn);
        mrow[e] = mn;
        const float p0 = __expf(s0 - mn);
        const float p1 = __expf(s1 - mn);
        float rs = p0 + p1;
        rs += __shfl_xor(rs, 1, 32);
        rs += __shfl_xor(rs, 2, 32);
        rs += __shfl_xor(rs, 4, 32);
        rs += __shfl_xor(rs, 8, 32);
        lrow[e] = lrow[e] * al + rs;
#pragma unroll
        for (int jd = 0; jd < 4; ++jd) o[jd][e] *= al;
        pw[(e + half * 8) * PLD + lm]      = f2bf(p0);  // C-frag -> LDS
        pw[(e + half * 8) * PLD + 16 + lm] = f2bf(p1);
      }
      // reload P as an A fragment (transpose through wave-private LDS)
      Frag ap;
      const u16* pb = &pw[lm * PLD + half * 8];
      ap.h[0] = *(const v8bf*)pb;
      ap.h[1] = *(const v8bf*)(pb + 16);
#pragma unroll
      for (int jd = 0; jd < 4; ++jd) {       // B = V chunk: lane = d col
        Frag bv;
        const u16* vp = &Vt[cur][(jd * 16 + lm) * VLD + half * 8];
        bv.h[0] = *(const v8bf*)vp;
        bv.h[1] = *(const v8bf*)(vp + 16);
        o[jd] = wmma_bf16(ap, bv, o[jd]);
      }
    }
    wait_async0();
    __syncthreads();
  }

  // normalize + write y[b][t][h*64+d] bf16
  const int bb = bh >> 4, hh = bh & 15;
#pragma unroll
  for (int e = 0; e < 8; ++e) {
    const float inv = 1.0f / lrow[e];
    const int t = qbase + e + half * 8;
    u16* yr = yb + ((size_t)(bb * NT + t)) * NE + hh * ND;
#pragma unroll
    for (int jd = 0; jd < 4; ++jd) yr[jd * 16 + lm] = f2bf(o[jd][e] * inv);
  }
}

// ---------------------------------------------------------------------------
// Kernel 3: output projection GEMM (8192x1024x1024) + bias, fp32 out
// ---------------------------------------------------------------------------
__global__ __launch_bounds__(256) void k_outproj(
    const u16* __restrict__ yb,      // [8192,1024] bf16
    const u16* __restrict__ wb,      // [1024,1024] bf16
    const float* __restrict__ bias,  // [1024]
    float* __restrict__ out) {       // [8192,1024] fp32
  __shared__ __align__(16) u16 At[2][128 * LDT];
  __shared__ __align__(16) u16 Bt[2][128 * LDT];
  const int tid  = threadIdx.x;
  const int lane = tid & 31, wid = tid >> 5;
  const int half = lane >> 4, lm = lane & 15;
  const int wm = wid >> 2, wn = wid & 3;
  const int m0 = blockIdx.y * 128, n0 = blockIdx.x * 128;

  v8f acc[4][2];
#pragma unroll
  for (int i = 0; i < 4; ++i)
#pragma unroll
    for (int j = 0; j < 2; ++j) acc[i][j] = vzero8();

#pragma unroll
  for (int p = 0; p < 2; ++p) {
    int id = tid + p * 256;
    int rr = id >> 2, cc = (id & 3) * 8;
    async_ld16(yb + (size_t)(m0 + rr) * 1024 + cc, &At[0][rr * LDT + cc]);
    async_ld16(wb + (size_t)(n0 + rr) * 1024 + cc, &Bt[0][rr * LDT + cc]);
  }
  wait_async0();
  __syncthreads();

  int cur = 0;
  for (int k0 = 0; k0 < 1024; k0 += 32, cur ^= 1) {
    if (k0 + 32 < 1024) {
#pragma unroll
      for (int p = 0; p < 2; ++p) {
        int id = tid + p * 256;
        int rr = id >> 2, cc = (id & 3) * 8;
        async_ld16(yb + (size_t)(m0 + rr) * 1024 + k0 + 32 + cc,
                   &At[cur ^ 1][rr * LDT + cc]);
        async_ld16(wb + (size_t)(n0 + rr) * 1024 + k0 + 32 + cc,
                   &Bt[cur ^ 1][rr * LDT + cc]);
      }
    }
    Frag a[4], b[2];
#pragma unroll
    for (int i = 0; i < 4; ++i) {
      const u16* ab = &At[cur][(wm * 64 + i * 16 + lm) * LDT + half * 8];
      a[i].h[0] = *(const v8bf*)ab;
      a[i].h[1] = *(const v8bf*)(ab + 16);
    }
#pragma unroll
    for (int j = 0; j < 2; ++j) {
      const u16* bb = &Bt[cur][(wn * 32 + j * 16 + lm) * LDT + half * 8];
      b[j].h[0] = *(const v8bf*)bb;
      b[j].h[1] = *(const v8bf*)(bb + 16);
    }
#pragma unroll
    for (int i = 0; i < 4; ++i)
#pragma unroll
      for (int j = 0; j < 2; ++j) acc[i][j] = wmma_bf16(a[i], b[j], acc[i][j]);
    wait_async0();
    __syncthreads();
  }
#pragma unroll
  for (int j = 0; j < 2; ++j) {
    const int n = n0 + wn * 32 + j * 16 + lm;
    const float bn = bias[n];
#pragma unroll
    for (int i = 0; i < 4; ++i)
#pragma unroll
      for (int e = 0; e < 8; ++e) {
        const int m = m0 + wm * 64 + i * 16 + e + half * 8;
        out[(size_t)m * 1024 + n] = acc[i][j][e] + bn;
      }
  }
}

// ---------------------------------------------------------------------------
// Host launcher
// ---------------------------------------------------------------------------
extern "C" void kernel_launch(void* const* d_in, const int* in_sizes, int n_in,
                              void* d_out, int out_size, void* d_ws, size_t ws_size,
                              hipStream_t stream) {
  const float* x     = (const float*)d_in[0];   // [8,1024,1024]
  const float* w_qkv = (const float*)d_in[1];   // [3072,1024]
  const float* b_qkv = (const float*)d_in[2];   // [3072]
  const float* w_out = (const float*)d_in[3];   // [1024,1024]
  const float* b_out = (const float*)d_in[4];   // [1024]
  const float* cost  = (const float*)d_in[5];   // [4095,32]
  const float* sint  = (const float*)d_in[6];   // [4095,32]
  float* out = (float*)d_out;

  char* ws = (char*)d_ws;
  const size_t MB = 1024 * 1024;
  u16* xb   = (u16*)(ws + 0 * MB);    // 16 MB  x bf16
  u16* wqkb = (u16*)(ws + 16 * MB);   //  6 MB  w_qkv bf16
  u16* wob  = (u16*)(ws + 22 * MB);   //  2 MB  w_out bf16
  u16* qo   = (u16*)(ws + 24 * MB);   // 16 MB  q [B,H,T,D]
  u16* ko   = (u16*)(ws + 40 * MB);   // 16 MB  k [B,H,T,D]
  u16* vto  = (u16*)(ws + 56 * MB);   // 16 MB  v [B,H,D,T]
  u16* ybuf = (u16*)(ws + 72 * MB);   // 16 MB  y [B,T,E]      (total 88 MB)

  k_cvt_bf16<<<8192, 256, 0, stream>>>(x,     xb,   8 * 1024 * 1024);
  k_cvt_bf16<<<3072, 256, 0, stream>>>(w_qkv, wqkb, 3 * 1024 * 1024);
  k_cvt_bf16<<<1024, 256, 0, stream>>>(w_out, wob,  1 * 1024 * 1024);

  k_qkv_rope<<<dim3(24, 64), 256, 0, stream>>>(xb, wqkb, b_qkv, cost, sint,
                                               qo, ko, vto);
  k_attn<<<dim3(NB * NH, NT / 128), 256, 0, stream>>>(qo, ko, vto, ybuf);
  k_outproj<<<dim3(8, 64), 256, 0, stream>>>(ybuf, wob, b_out, out);
}
